// MultiHeadAttention_66348654788798
// MI455X (gfx1250) — matrix operand order repaired and checked by reading, not compiled, not run on hardware
//
#include <hip/hip_runtime.h>
#include <hip/hip_bf16.h>

// ---------- types ----------
typedef __attribute__((ext_vector_type(16))) __bf16 v16bf;
typedef __attribute__((ext_vector_type(8)))  float  v8f;
typedef __attribute__((ext_vector_type(4)))  unsigned int u32x4;
typedef __attribute__((ext_vector_type(8)))  int i32x8;
typedef __attribute__((ext_vector_type(4)))  int i32x4;

union BF16x16 { v16bf v; uint4 q[2]; };
union U4S8    { uint4 u; unsigned short s[8]; };

__device__ __forceinline__ unsigned short f2bf(float f) {
  union { float f; unsigned u; } x; x.f = f;
  unsigned r = x.u + 0x7FFFu + ((x.u >> 16) & 1u);   // round-to-nearest-even
  return (unsigned short)(r >> 16);
}

#define HAS_TDM __has_builtin(__builtin_amdgcn_tensor_load_to_lds)

#if HAS_TDM
// TDM descriptor group 0: count=1, lds addr, 57-bit global addr, type=2
__device__ __forceinline__ u32x4 tdm_g0(unsigned lds_off, const void* gaddr) {
  unsigned long long a = (unsigned long long)gaddr;
  u32x4 g;
  g[0] = 1u;                                               // count=1, user mode
  g[1] = lds_off;                                          // lds_addr (bytes)
  g[2] = (unsigned)(a & 0xFFFFFFFFu);                      // global_addr[31:0]
  g[3] = (unsigned)((a >> 32) & 0x01FFFFFFu) | (2u << 30); // global_addr[56:32] | type=2
  return g;
}

// TDM descriptor group 1: 2-D tile tile0 x tile1 (2-byte elements), row stride
// `stride` elements, LDS pad of (1<<(padi+1)) DWORDs interval, (pada+1) DWORDs pad.
__device__ __forceinline__ i32x8 tdm_g1(int dim0, int dim1, int tile0, int tile1,
                                        int stride, int padi, int pada) {
  i32x8 g;
  g[0] = (1 << 16) |            // data_size = 2 bytes
         (1 << 20) |            // pad_enable
         (padi << 22) |         // pad_interval
         (pada << 25);          // pad_amount
  g[1] = (int)((unsigned)(dim0 & 0xFFFF) << 16);             // tensor_dim0[15:0]
  g[2] = (int)(((unsigned)dim0 >> 16) |
               ((unsigned)(dim1 & 0xFFFF) << 16));           // dim0 hi | tensor_dim1 lo
  g[3] = (int)(((unsigned)dim1 >> 16) |
               ((unsigned)tile0 << 16));                     // dim1 hi | tile_dim0
  g[4] = tile1;                                              // tile_dim1 (tile_dim2 = 0)
  g[5] = stride;                                             // tensor_dim0_stride[31:0]
  g[6] = 0;
  g[7] = 0;
  return g;
}
#endif

// ---------- pack kernels ----------
__global__ __launch_bounds__(256)
void f32_to_bf16_kernel(const float* __restrict__ src, unsigned short* __restrict__ dst, int n) {
  int i = blockIdx.x * 256 + threadIdx.x;
  if (i < n) dst[i] = f2bf(src[i]);
}

// dst[n*K + k] = bf16(src[k*N + n])   (W [K][N] -> W^T [N][K])
__global__ __launch_bounds__(256)
void transpose_bf16_kernel(const float* __restrict__ src, unsigned short* __restrict__ dst,
                           int K, int N) {
  int i = blockIdx.x * 256 + threadIdx.x;
  if (i < K * N) {
    int k = i / N, n = i - k * N;
    dst[(size_t)n * K + k] = f2bf(src[i]);
  }
}

// ---------- GEMM: C[M x N] = A[M x K](bf16) * BT[N x K](bf16)^T + bias ----------
// 128 threads = 4 waves; block tile 128x128; wave tile 64x64 (16 WMMA / 16 ds_load per iter)
#define BM 128
#define BN 128
#define BKK 32
#define PK 40   // padded LDS pitch (half-words): 32 data + 8 pad (TDM: 16 dw + 4 dw pad)

__global__ __launch_bounds__(128)
void gemm_bf16_kernel(const unsigned short* __restrict__ A,
                      const unsigned short* __restrict__ BT,
                      const float* __restrict__ bias,
                      unsigned short* __restrict__ outB,   // bf16 out (or null)
                      float* __restrict__ outF,            // f32 out (or null)
                      int M, int N, int K)
{
  __shared__ __align__(16) unsigned short As[2][BM * PK];
  __shared__ __align__(16) unsigned short Bs[2][BN * PK];

  const int tid  = threadIdx.x;
  const int lane = tid & 31;
  const int wave = tid >> 5;        // 0..3
  const int half = lane >> 4;
  const int lr   = lane & 15;
  const int wm   = wave >> 1;       // 0..1
  const int wn   = wave & 1;        // 0..1
  const int bm   = blockIdx.y * BM;
  const int bn   = blockIdx.x * BN;

  v8f acc[4][4];
#pragma unroll
  for (int ms = 0; ms < 4; ++ms)
#pragma unroll
    for (int ns = 0; ns < 4; ++ns) acc[ms][ns] = (v8f)(0.0f);

  const int iters = K / BKK;

#if HAS_TDM
  // ---- Tensor Data Mover path: DMA tiles global->LDS, double-buffered ----
  const i32x4 z4 = {0, 0, 0, 0};
  const i32x8 z8 = {0, 0, 0, 0, 0, 0, 0, 0};
  const i32x8 g1A = tdm_g1(K, M, BKK, BM, K, 3, 3);   // 16-dw rows + 4-dw pad -> pitch 40 hw
  const i32x8 g1B = tdm_g1(K, N, BKK, BN, K, 3, 3);
  const unsigned ldsA[2] = { (unsigned)(unsigned long long)&As[0][0],
                             (unsigned)(unsigned long long)&As[1][0] };
  const unsigned ldsB[2] = { (unsigned)(unsigned long long)&Bs[0][0],
                             (unsigned)(unsigned long long)&Bs[1][0] };
  const unsigned short* gA = A  + (size_t)bm * K;
  const unsigned short* gB = BT + (size_t)bn * K;

  if (wave == 0) {
    __builtin_amdgcn_tensor_load_to_lds(tdm_g0(ldsA[0], gA), g1A, z4, z4, z8, 0);
    __builtin_amdgcn_tensor_load_to_lds(tdm_g0(ldsB[0], gB), g1B, z4, z4, z8, 0);
  }
  for (int it = 0; it < iters; ++it) {
    if (wave == 0) {
      if (it + 1 < iters) {
        int kn = (it + 1) * BKK;
        __builtin_amdgcn_tensor_load_to_lds(tdm_g0(ldsA[(it + 1) & 1], gA + kn), g1A, z4, z4, z8, 0);
        __builtin_amdgcn_tensor_load_to_lds(tdm_g0(ldsB[(it + 1) & 1], gB + kn), g1B, z4, z4, z8, 0);
        __builtin_amdgcn_s_wait_tensorcnt(2);   // current buffer's 2 DMAs done
      } else {
        __builtin_amdgcn_s_wait_tensorcnt(0);
      }
    }
    __syncthreads();   // publish DMA'd tile to all waves

    const unsigned short* Asb = &As[it & 1][0];
    const unsigned short* Bsb = &Bs[it & 1][0];

    BF16x16 afr[4], bfr[4];
#pragma unroll
    for (int ms = 0; ms < 4; ++ms) {
      int row = wm * 64 + ms * 16 + lr;
      afr[ms].q[0] = *(const uint4*)(Asb + row * PK + half * 8);
      afr[ms].q[1] = *(const uint4*)(Asb + row * PK + 16 + half * 8);
    }
#pragma unroll
    for (int ns = 0; ns < 4; ++ns) {
      int col = wn * 64 + ns * 16 + lr;
      bfr[ns].q[0] = *(const uint4*)(Bsb + col * PK + half * 8);
      bfr[ns].q[1] = *(const uint4*)(Bsb + col * PK + 16 + half * 8);
    }
#pragma unroll
    for (int ms = 0; ms < 4; ++ms)
#pragma unroll
      for (int ns = 0; ns < 4; ++ns)
        acc[ms][ns] = __builtin_amdgcn_wmma_f32_16x16x32_bf16(
            false, afr[ms].v, false, bfr[ns].v, (short)0, acc[ms][ns], false, false);

    __syncthreads();   // all waves done reading before buffer is DMA'd again
  }
#else
  // ---- fallback: per-thread staged copy through VGPRs ----
  for (int it = 0; it < iters; ++it) {
    int k0 = it * BKK;
    uint4 av[4], bv[4];
#pragma unroll
    for (int c = 0; c < 4; ++c) {
      av[c] = *(const uint4*)(A  + (size_t)(bm + tid) * K + k0 + c * 8);
      bv[c] = *(const uint4*)(BT + (size_t)(bn + tid) * K + k0 + c * 8);
    }
    __syncthreads();
#pragma unroll
    for (int c = 0; c < 4; ++c) {
      *(uint4*)(&As[0][0] + tid * PK + c * 8) = av[c];
      *(uint4*)(&Bs[0][0] + tid * PK + c * 8) = bv[c];
    }
    __syncthreads();

    BF16x16 afr[4], bfr[4];
#pragma unroll
    for (int ms = 0; ms < 4; ++ms) {
      int row = wm * 64 + ms * 16 + lr;
      afr[ms].q[0] = *(const uint4*)(&As[0][0] + row * PK + half * 8);
      afr[ms].q[1] = *(const uint4*)(&As[0][0] + row * PK + 16 + half * 8);
    }
#pragma unroll
    for (int ns = 0; ns < 4; ++ns) {
      int col = wn * 64 + ns * 16 + lr;
      bfr[ns].q[0] = *(const uint4*)(&Bs[0][0] + col * PK + half * 8);
      bfr[ns].q[1] = *(const uint4*)(&Bs[0][0] + col * PK + 16 + half * 8);
    }
#pragma unroll
    for (int ms = 0; ms < 4; ++ms)
#pragma unroll
      for (int ns = 0; ns < 4; ++ns)
        acc[ms][ns] = __builtin_amdgcn_wmma_f32_16x16x32_bf16(
            false, afr[ms].v, false, bfr[ns].v, (short)0, acc[ms][ns], false, false);
  }
#endif

  // epilogue: C layout -> VGPR i holds (M = i + half*8, N = lr)
#pragma unroll
  for (int ms = 0; ms < 4; ++ms) {
#pragma unroll
    for (int ns = 0; ns < 4; ++ns) {
      int gm0 = bm + wm * 64 + ms * 16 + half * 8;
      int gn  = bn + wn * 64 + ns * 16 + lr;
      float bb = bias[gn];
#pragma unroll
      for (int i = 0; i < 8; ++i) {
        float v = acc[ms][ns][i] + bb;
        size_t idx = (size_t)(gm0 + i) * N + gn;
        if (outB) outB[idx] = f2bf(v);
        else      outF[idx] = v;
      }
    }
  }
}

// ---------- flash attention: Q,K,V bf16 [4096][1024], per (b,h) slice stride 64 ----------
#define KP 72   // K tile pitch (half-words), 32 rows x 64 d (TDM: 32 dw + 4 dw pad)
#define VP 40   // V^T pitch, 64 d rows x 32 keys
#define PP 40   // P stage pitch, per-wave 16 x 32

__global__ __launch_bounds__(256)
void attn_kernel(const unsigned short* __restrict__ Q,
                 const unsigned short* __restrict__ Km,
                 const unsigned short* __restrict__ Vm,
                 unsigned short* __restrict__ Ob)
{
  __shared__ __align__(16) unsigned short Ks[2][32 * KP];
  __shared__ __align__(16) unsigned short Vt[64 * VP];
  __shared__ __align__(16) unsigned short Ps[8][16 * PP];

  const int tid  = threadIdx.x;
  const int lane = tid & 31;
  const int wave = tid >> 5;
  const int half = lane >> 4;
  const int lr   = lane & 15;

  const int bh = blockIdx.y;
  const int b  = bh >> 4;
  const int h  = bh & 15;
  const int q0 = blockIdx.x * 128 + wave * 16;

  const unsigned short* qbase = Q  + (size_t)b * 2048 * 1024 + h * 64;
  const unsigned short* kbase = Km + (size_t)b * 2048 * 1024 + h * 64;
  const unsigned short* vbase = Vm + (size_t)b * 2048 * 1024 + h * 64;

  // Q fragments (kept in registers for the whole sweep)
  BF16x16 qf[2];
#pragma unroll
  for (int c = 0; c < 2; ++c) {
    int row = q0 + lr;
    int d0  = c * 32 + half * 8;
    qf[c].q[0] = *(const uint4*)(qbase + (size_t)row * 1024 + d0);
    qf[c].q[1] = *(const uint4*)(qbase + (size_t)row * 1024 + d0 + 16);
  }

  float m_i[8], l_i[8];
  v8f o[4];
#pragma unroll
  for (int i = 0; i < 8; ++i) { m_i[i] = -1e30f; l_i[i] = 0.0f; }
#pragma unroll
  for (int d = 0; d < 4; ++d) o[d] = (v8f)(0.0f);

  const int krow = tid >> 3;          // 0..31
  const int kc   = (tid & 7) * 8;     // 0..56 (half-words inside Dk=64)
  const int iters = 2048 / 32;

#if HAS_TDM
  const i32x4 z4 = {0, 0, 0, 0};
  const i32x8 z8 = {0, 0, 0, 0, 0, 0, 0, 0};
  // K tile: 64 x 32 elements, stride 1024; 32-dw rows + 4-dw pad -> pitch 72 hw
  const i32x8 g1K = tdm_g1(1024, 4096, 64, 32, 1024, 4, 3);
  const unsigned ldsK[2] = { (unsigned)(unsigned long long)&Ks[0][0],
                             (unsigned)(unsigned long long)&Ks[1][0] };
  if (wave == 0)
    __builtin_amdgcn_tensor_load_to_lds(tdm_g0(ldsK[0], kbase), g1K, z4, z4, z8, 0);
#endif

  for (int it = 0; it < iters; ++it) {
    const int kb0 = it * 32;
#if !HAS_TDM
    uint4 kv = *(const uint4*)(kbase + (size_t)(kb0 + krow) * 1024 + kc);
#endif
    uint4 vv = *(const uint4*)(vbase + (size_t)(kb0 + krow) * 1024 + kc);

    __syncthreads();   // previous iteration done with Ks/Vt
#if HAS_TDM
    if (wave == 0) {
      if (it + 1 < iters)
        __builtin_amdgcn_tensor_load_to_lds(
            tdm_g0(ldsK[(it + 1) & 1], kbase + (size_t)(kb0 + 32) * 1024), g1K, z4, z4, z8, 0);
    }
#else
    *(uint4*)(&Ks[0][0] + krow * KP + kc) = kv;
#endif
    U4S8 u; u.u = vv;
#pragma unroll
    for (int j = 0; j < 8; ++j)
      Vt[(kc + j) * VP + krow] = u.s[j];   // transposed store: [d][key]
#if HAS_TDM
    if (wave == 0) {
      if (it + 1 < iters) __builtin_amdgcn_s_wait_tensorcnt(1);  // tile `it` done
      else                __builtin_amdgcn_s_wait_tensorcnt(0);
    }
    const unsigned short* Ksb = &Ks[it & 1][0];
#else
    const unsigned short* Ksb = &Ks[0][0];
#endif
    __syncthreads();

    // scores: S = Q * K^T  (two 16-key tiles, K-dim = 64 in two chunks)
    v8f s[2];
    s[0] = (v8f)(0.0f); s[1] = (v8f)(0.0f);
#pragma unroll
    for (int ns = 0; ns < 2; ++ns) {
      int key = ns * 16 + lr;
#pragma unroll
      for (int c = 0; c < 2; ++c) {
        BF16x16 bf;
        bf.q[0] = *(const uint4*)(Ksb + key * KP + c * 32 + half * 8);
        bf.q[1] = *(const uint4*)(Ksb + key * KP + c * 32 + 16 + half * 8);
        s[ns] = __builtin_amdgcn_wmma_f32_16x16x32_bf16(
            false, qf[c].v, false, bf.v, (short)0, s[ns], false, false);
      }
    }
#pragma unroll
    for (int ns = 0; ns < 2; ++ns)
#pragma unroll
      for (int i = 0; i < 8; ++i) s[ns][i] *= 0.125f;   // 1/sqrt(64)

    // online softmax (row reductions across the 16-lane half)
    float mnew[8], alpha[8];
#pragma unroll
    for (int i = 0; i < 8; ++i) {
      float t = fmaxf(s[0][i], s[1][i]);
      t = fmaxf(t, __shfl_xor(t, 1));
      t = fmaxf(t, __shfl_xor(t, 2));
      t = fmaxf(t, __shfl_xor(t, 4));
      t = fmaxf(t, __shfl_xor(t, 8));
      mnew[i]  = fmaxf(m_i[i], t);
      alpha[i] = __expf(m_i[i] - mnew[i]);
      m_i[i]   = mnew[i];
    }
#pragma unroll
    for (int i = 0; i < 8; ++i) {
      float p0 = __expf(s[0][i] - mnew[i]);
      float p1 = __expf(s[1][i] - mnew[i]);
      s[0][i] = p0; s[1][i] = p1;
      float r = p0 + p1;
      r += __shfl_xor(r, 1);
      r += __shfl_xor(r, 2);
      r += __shfl_xor(r, 4);
      r += __shfl_xor(r, 8);
      l_i[i] = l_i[i] * alpha[i] + r;
    }
#pragma unroll
    for (int d = 0; d < 4; ++d)
#pragma unroll
      for (int i = 0; i < 8; ++i) o[d][i] *= alpha[i];

    // P: C-layout -> A-fragment layout via per-wave LDS stage
    unsigned short* pw = &Ps[wave][0];
#pragma unroll
    for (int ns = 0; ns < 2; ++ns)
#pragma unroll
      for (int i = 0; i < 8; ++i)
        pw[(i + half * 8) * PP + ns * 16 + lr] = f2bf(s[ns][i]);
    asm volatile("s_wait_dscnt 0" ::: "memory");   // same-wave LDS RAW, in-order

    BF16x16 pf;
    pf.q[0] = *(const uint4*)(pw + lr * PP + half * 8);
    pf.q[1] = *(const uint4*)(pw + lr * PP + 16 + half * 8);

    // O += P * V  (B fragment from transposed V tile)
#pragma unroll
    for (int ds = 0; ds < 4; ++ds) {
      int d = ds * 16 + lr;
      BF16x16 vf;
      vf.q[0] = *(const uint4*)(Vt + d * VP + half * 8);
      vf.q[1] = *(const uint4*)(Vt + d * VP + 16 + half * 8);
      o[ds] = __builtin_amdgcn_wmma_f32_16x16x32_bf16(
          false, pf.v, false, vf.v, (short)0, o[ds], false, false);
    }
  }

  // normalize and emit bf16 for the output projection
#pragma unroll
  for (int i = 0; i < 8; ++i) {
    float inv = 1.0f / l_i[i];
    int row = q0 + i + half * 8;
#pragma unroll
    for (int ds = 0; ds < 4; ++ds) {
      int d = ds * 16 + lr;
      Ob[(size_t)(b * 2048 + row) * 1024 + h * 64 + d] = f2bf(o[ds][i] * inv);
    }
  }
}

// ---------- host ----------
extern "C" void kernel_launch(void* const* d_in, const int* in_sizes, int n_in,
                              void* d_out, int out_size, void* d_ws, size_t ws_size,
                              hipStream_t stream) {
  const float* x  = (const float*)d_in[0];
  const float* Wq = (const float*)d_in[1];
  const float* bq = (const float*)d_in[2];
  const float* Wk = (const float*)d_in[3];
  const float* bk = (const float*)d_in[4];
  const float* Wv = (const float*)d_in[5];
  const float* bv = (const float*)d_in[6];
  const float* Wo = (const float*)d_in[7];
  const float* bo = (const float*)d_in[8];
  float* out = (float*)d_out;

  char* w = (char*)d_ws;
  const size_t MB = 1u << 20;
  unsigned short* xb  = (unsigned short*)(w + 0);        // 4096x1024 bf16 (8 MB)
  unsigned short* wqT = (unsigned short*)(w + 8  * MB);  // 1024x1024 bf16 (2 MB)
  unsigned short* wkT = (unsigned short*)(w + 10 * MB);
  unsigned short* wvT = (unsigned short*)(w + 12 * MB);
  unsigned short* woT = (unsigned short*)(w + 14 * MB);
  unsigned short* qb  = (unsigned short*)(w + 16 * MB);  // 8 MB each
  unsigned short* kb  = (unsigned short*)(w + 24 * MB);
  unsigned short* vb  = (unsigned short*)(w + 32 * MB);
  unsigned short* ab  = (unsigned short*)(w + 40 * MB);

  const int M = 4096, N = 1024, K = 1024;

  f32_to_bf16_kernel<<<(M * K) / 256, 256, 0, stream>>>(x, xb, M * K);
  transpose_bf16_kernel<<<(K * N) / 256, 256, 0, stream>>>(Wq, wqT, K, N);
  transpose_bf16_kernel<<<(K * N) / 256, 256, 0, stream>>>(Wk, wkT, K, N);
  transpose_bf16_kernel<<<(K * N) / 256, 256, 0, stream>>>(Wv, wvT, K, N);
  transpose_bf16_kernel<<<(K * N) / 256, 256, 0, stream>>>(Wo, woT, K, N);

  dim3 ggrid(N / BN, M / BM);
  gemm_bf16_kernel<<<ggrid, 128, 0, stream>>>(xb, wqT, bq, qb, nullptr, M, N, K);
  gemm_bf16_kernel<<<ggrid, 128, 0, stream>>>(xb, wkT, bk, kb, nullptr, M, N, K);
  gemm_bf16_kernel<<<ggrid, 128, 0, stream>>>(xb, wvT, bv, vb, nullptr, M, N, K);

  dim3 agrid(2048 / 128, 32);
  attn_kernel<<<agrid, 256, 0, stream>>>(qb, kb, vb, ab);

  gemm_bf16_kernel<<<ggrid, 128, 0, stream>>>(ab, woT, bo, nullptr, out, M, N, K);
}
